// AdaLanczosNet_23398981828720
// MI455X (gfx1250) — compile-verified, matrix operand order checked by reading, not run on hardware
//
#include <hip/hip_runtime.h>

// ---------------------------------------------------------------------------
// AdaLanczosNet on gfx1250 (MI455X). All dense contractions go through a
// wave32 WMMA bf16 GEMM (f32 accumulate): 4 waves / 32x64 macro-tile, each
// wave a 16x32 sub-tile (2 accumulators), K chunked by 64 -> 4 v_wmma sites.
// Fragments are contiguous in LDS (ds_load_b128 x2 per fragment). Sequential
// Lanczos / T-power recurrences run as one workgroup per batch with LDS +
// deterministic tree reductions. No atomics -> deterministic across replays.
// ---------------------------------------------------------------------------

#define BB 16
#define NN 256
#define DA 64      // NUM_ATOM
#define KK 20
#define EPSF 1.1920929e-07f
#define LBF 1e-4f

typedef __attribute__((ext_vector_type(16))) __bf16 v16bf;
typedef __attribute__((ext_vector_type(8)))  __bf16 v8bf;
typedef __attribute__((ext_vector_type(8)))  float  v8f;

#define LDK 72   // LDS row stride in bf16: 144B = 36 banks -> conflict-free

__device__ __forceinline__ __bf16 f2bf(float f) {
  unsigned u = __builtin_bit_cast(unsigned, f);
  unsigned r = (u + 0x7FFFu + ((u >> 16) & 1u)) >> 16;   // RNE
  return __builtin_bit_cast(__bf16, (unsigned short)r);
}

__device__ __forceinline__ unsigned wang(unsigned s) {
  s = (s ^ 61u) ^ (s >> 16); s *= 9u; s ^= s >> 4; s *= 0x27d4eb2du; s ^= s >> 15;
  return s;
}
__device__ __forceinline__ float u01(unsigned s) {
  return 1.0f - (wang(s) >> 8) * (1.0f / 16777216.0f);   // (0,1]
}
__device__ __forceinline__ float nrand(unsigned s) {
  float u1 = u01(s * 2u + 1u), u2 = u01(s * 2u + 2u);
  return sqrtf(-2.0f * logf(u1)) * cosf(6.28318530718f * u2);
}

__device__ __forceinline__ float blockReduce256(float v, float* red) {
  int t = threadIdx.x;
  red[t] = v; __syncthreads();
  for (int off = 128; off > 0; off >>= 1) {
    if (t < off) red[t] += red[t + off];
    __syncthreads();
  }
  float r = red[0]; __syncthreads();
  return r;
}

// ---------------------------------------------------------------------------
// Batched GEMM: C[b] = op(A[b]) * op(B[b]) (+bias)(+relu).
// Block = 128 threads (4 waves). Macro-tile 32(M) x 64(N); wave w owns rows
// (w>>1)*16.. and cols (w&1)*32.. (two 16x16 accumulators). K chunk = 64.
// A staged as As[m][k]; B staged TRANSPOSED as Bsw[n][k] so each lane's
// fragment is two contiguous 16B runs -> ds_load_b128.
// ---------------------------------------------------------------------------
__global__ void k_wmma_gemm(const float* __restrict__ A, int lda, long long sA,
                            const float* __restrict__ Bm, int ldb, long long sB,
                            float* __restrict__ C, int ldc, long long sC,
                            int M, int N, int K,
                            const float* __restrict__ bias,
                            int relu, int tA, int tB) {
  __shared__ __align__(16) __bf16 As[32][LDK];   // 32 m-rows x 64 k
  __shared__ __align__(16) __bf16 Bsw[64][LDK];  // 64 n-cols x 64 k

  const int tid  = threadIdx.x;        // 0..127
  const int lane = tid & 31;
  const int w    = tid >> 5;           // wave id 0..3
  const int wm   = (w >> 1) << 4;      // wave row offset (0/16)
  const int wn   = (w & 1) << 5;       // wave col offset (0/32)
  const int hl   = lane >> 4;          // K-half select within fragment
  const int mr   = lane & 15;          // row (A) / col (B) within 16x16

  const int bz = blockIdx.z;
  const int m0 = blockIdx.y * 32;
  const int n0 = blockIdx.x * 64;
  const float* Ab = A + (long long)bz * sA;
  const float* Bb = Bm + (long long)bz * sB;
  float* Cb = C + (long long)bz * sC;

  // loader assignments
  const int ar  = tid >> 2;            // A m-row 0..31
  const int ac0 = (tid & 3) << 4;      // A k segment (16 wide)
  const int bn  = tid & 63;            // B n-col 0..63
  const int bks = (tid >> 6) << 5;     // B k segment (32 wide): 0 or 32

  v8f acc0 = {}, acc1 = {};

  for (int kc = 0; kc < K; kc += 64) {
    // ---- stage A tile (32 x 64) ----
    {
      int gm = m0 + ar;
      v8bf lo = {}, hi = {};
      if (!tA && gm < M && (kc + ac0 + 16) <= K) {
        const float* src = Ab + (long long)gm * lda + kc + ac0;
#pragma unroll
        for (int u = 0; u < 8; u += 4) {
          float4 v4 = *(const float4*)(src + u);
          lo[u + 0] = f2bf(v4.x); lo[u + 1] = f2bf(v4.y);
          lo[u + 2] = f2bf(v4.z); lo[u + 3] = f2bf(v4.w);
          float4 w4 = *(const float4*)(src + 8 + u);
          hi[u + 0] = f2bf(w4.x); hi[u + 1] = f2bf(w4.y);
          hi[u + 2] = f2bf(w4.z); hi[u + 3] = f2bf(w4.w);
        }
      } else {
#pragma unroll
        for (int u = 0; u < 8; ++u) {
          int gk0 = kc + ac0 + u, gk1 = gk0 + 8;
          float v0 = 0.0f, v1 = 0.0f;
          if (gm < M && gk0 < K)
            v0 = tA ? Ab[(long long)gk0 * lda + gm] : Ab[(long long)gm * lda + gk0];
          if (gm < M && gk1 < K)
            v1 = tA ? Ab[(long long)gk1 * lda + gm] : Ab[(long long)gm * lda + gk1];
          lo[u] = f2bf(v0); hi[u] = f2bf(v1);
        }
      }
      *(v8bf*)&As[ar][ac0]     = lo;
      *(v8bf*)&As[ar][ac0 + 8] = hi;
    }
    // ---- stage B tile transposed: Bsw[n][k], 64 n x 64 k ----
    {
      int gn = bn + n0;
      v8bf q0 = {}, q1 = {}, q2 = {}, q3 = {};
      if (gn < N && (kc + bks + 32) <= K) {
        if (tB) {   // B[n][k] row-major: contiguous k run
          const float* src = Bb + (long long)gn * ldb + kc + bks;
#pragma unroll
          for (int u = 0; u < 8; u += 4) {
            float4 a4 = *(const float4*)(src + u);
            q0[u+0]=f2bf(a4.x); q0[u+1]=f2bf(a4.y); q0[u+2]=f2bf(a4.z); q0[u+3]=f2bf(a4.w);
            float4 b4 = *(const float4*)(src + 8 + u);
            q1[u+0]=f2bf(b4.x); q1[u+1]=f2bf(b4.y); q1[u+2]=f2bf(b4.z); q1[u+3]=f2bf(b4.w);
            float4 c4 = *(const float4*)(src + 16 + u);
            q2[u+0]=f2bf(c4.x); q2[u+1]=f2bf(c4.y); q2[u+2]=f2bf(c4.z); q2[u+3]=f2bf(c4.w);
            float4 d4 = *(const float4*)(src + 24 + u);
            q3[u+0]=f2bf(d4.x); q3[u+1]=f2bf(d4.y); q3[u+2]=f2bf(d4.z); q3[u+3]=f2bf(d4.w);
          }
        } else {    // B[k][n]: strided per thread, coalesced across the wave
          const float* src = Bb + (long long)(kc + bks) * ldb + gn;
#pragma unroll
          for (int u = 0; u < 8; ++u) {
            q0[u] = f2bf(src[(long long)(u)      * ldb]);
            q1[u] = f2bf(src[(long long)(u + 8)  * ldb]);
            q2[u] = f2bf(src[(long long)(u + 16) * ldb]);
            q3[u] = f2bf(src[(long long)(u + 24) * ldb]);
          }
        }
      } else {
#pragma unroll
        for (int u = 0; u < 8; ++u) {
          int k0 = kc + bks + u;
          float v0 = 0, v1 = 0, v2 = 0, v3 = 0;
          if (gn < N) {
            if (k0      < K) v0 = tB ? Bb[(long long)gn*ldb + k0]      : Bb[(long long)(k0)*ldb + gn];
            if (k0 + 8  < K) v1 = tB ? Bb[(long long)gn*ldb + k0 + 8]  : Bb[(long long)(k0+8)*ldb + gn];
            if (k0 + 16 < K) v2 = tB ? Bb[(long long)gn*ldb + k0 + 16] : Bb[(long long)(k0+16)*ldb + gn];
            if (k0 + 24 < K) v3 = tB ? Bb[(long long)gn*ldb + k0 + 24] : Bb[(long long)(k0+24)*ldb + gn];
          }
          q0[u] = f2bf(v0); q1[u] = f2bf(v1); q2[u] = f2bf(v2); q3[u] = f2bf(v3);
        }
      }
      *(v8bf*)&Bsw[bn][bks]      = q0;
      *(v8bf*)&Bsw[bn][bks + 8]  = q1;
      *(v8bf*)&Bsw[bn][bks + 16] = q2;
      *(v8bf*)&Bsw[bn][bks + 24] = q3;
    }
    __syncthreads();

    if (kc + 64 < K) {  // hint next A chunk -> global_prefetch_b8
      __builtin_prefetch(tA ? (Ab + (long long)(kc + 64) * lda + m0)
                            : (Ab + (long long)(m0 + wm + mr) * lda + kc + 64), 0, 1);
    }

    // ---- 2 sub-chunks x 2 col-tiles of 32-deep WMMA ----
    const __bf16* ap  = &As[wm + mr][0];
    const __bf16* bp0 = &Bsw[wn + mr][0];
    const __bf16* bp1 = &Bsw[wn + 16 + mr][0];
#pragma unroll
    for (int sub = 0; sub < 2; ++sub) {
      int k0 = (sub << 5) + (hl << 3);
      v8bf alo = *(const v8bf*)(ap + k0);
      v8bf ahi = *(const v8bf*)(ap + k0 + 16);
      v16bf af = __builtin_shufflevector(alo, ahi, 0,1,2,3,4,5,6,7,8,9,10,11,12,13,14,15);
      {
        v8bf blo = *(const v8bf*)(bp0 + k0);
        v8bf bhi = *(const v8bf*)(bp0 + k0 + 16);
        v16bf bf = __builtin_shufflevector(blo, bhi, 0,1,2,3,4,5,6,7,8,9,10,11,12,13,14,15);
        acc0 = __builtin_amdgcn_wmma_f32_16x16x32_bf16(
            false, af, false, bf, (short)0, acc0, false, false);
      }
      {
        v8bf blo = *(const v8bf*)(bp1 + k0);
        v8bf bhi = *(const v8bf*)(bp1 + k0 + 16);
        v16bf bf = __builtin_shufflevector(blo, bhi, 0,1,2,3,4,5,6,7,8,9,10,11,12,13,14,15);
        acc1 = __builtin_amdgcn_wmma_f32_16x16x32_bf16(
            false, af, false, bf, (short)0, acc1, false, false);
      }
    }
    __syncthreads();
  }

  // store D tiles: m_local = r + 8*hl, n_local = mr
#pragma unroll
  for (int part = 0; part < 2; ++part) {
    v8f a = part ? acc1 : acc0;
#pragma unroll
    for (int r = 0; r < 8; ++r) {
      int gm = m0 + wm + r + (hl << 3);
      int gn = n0 + wn + (part << 4) + mr;
      if (gm < M && gn < N) {
        float v = a[r];
        if (bias) v += bias[gn];
        if (relu) v = fmaxf(v, 0.0f);
        Cb[(long long)gm * ldc + gn] = v;
      }
    }
  }
}

// --------------------------- small elementwise kernels ---------------------
__global__ void k_emb1(const int* __restrict__ nf, const float* __restrict__ w1,
                       const float* __restrict__ b1, float* __restrict__ h1) {
  int id = blockIdx.x * 256 + threadIdx.x;
  if (id >= BB * NN * 1024) return;
  int bn = id >> 10, j = id & 1023;
  int atom = nf[bn];
  h1[id] = fmaxf(w1[atom * 1024 + j] + b1[j], 0.0f);
}

__global__ void k_onehot(const int* __restrict__ nf, float* __restrict__ st) {
  int id = blockIdx.x * 256 + threadIdx.x;
  if (id >= BB * NN * DA) return;
  int bn = id / DA, d = id % DA;
  st[id] = (nf[bn] == d) ? 1.0f : 0.0f;
}

__global__ void k_slice_edge(const float* __restrict__ L, float* __restrict__ le) {
  int id = blockIdx.x * 256 + threadIdx.x;
  if (id >= 4 * BB * NN * NN) return;
  int e = id / (BB * NN * NN);
  int r = id % (BB * NN * NN);
  le[id] = L[(long long)r * 4 + e];
}

__global__ void k_rownorm(const float* __restrict__ feat, float* __restrict__ nrm) {
  int id = blockIdx.x * 256 + threadIdx.x;
  if (id >= BB * NN) return;
  const float* row = feat + (long long)id * DA;
  float s = 0.0f;
  for (int d = 0; d < DA; ++d) s += row[d] * row[d];
  nrm[id] = s;
}

// in-place: G -> dist2, and per-row partial sums for sigma
__global__ void k_dist(float* __restrict__ g, const float* __restrict__ nrm,
                       float* __restrict__ rowpart) {
  __shared__ float red[256];
  int b = blockIdx.x / NN, i = blockIdx.x % NN, j = threadIdx.x;
  long long base = ((long long)b * NN + i) * NN;
  float d2 = nrm[b * NN + i] + nrm[b * NN + j] - 2.0f * g[base + j];
  g[base + j] = d2;
  float s = blockReduce256(d2, red);
  if (j == 0) rowpart[blockIdx.x] = s;
}

__global__ void k_sigma(const float* __restrict__ rowpart, float* __restrict__ sig) {
  __shared__ float red[256];
  int b = blockIdx.x;
  float s = blockReduce256(rowpart[b * NN + threadIdx.x], red);
  if (threadIdx.x == 0) sig[b] = s / (float)(NN * NN);
}

// dist2 -> A = exp(-d2/sigma)*mask  (in place), row sums of A
__global__ void k_affin(float* __restrict__ g, const float* __restrict__ L,
                        const float* __restrict__ sig, float* __restrict__ rowsum) {
  __shared__ float red[256];
  int b = blockIdx.x / NN, i = blockIdx.x % NN, j = threadIdx.x;
  long long base = ((long long)b * NN + i) * NN;
  float mask = L[(base + j) * 4 + 0];
  float a = expf(-g[base + j] / sig[b]) * mask;
  g[base + j] = a;
  float s = blockReduce256(a, red);
  if (j == 0) rowsum[blockIdx.x] = s;
}

__global__ void k_symnorm(float* __restrict__ g, const float* __restrict__ rowsum) {
  int b = blockIdx.x / NN, i = blockIdx.x % NN, j = threadIdx.x;
  long long base = ((long long)b * NN + i) * NN;
  float ri = rowsum[b * NN + i], rj = rowsum[b * NN + j];
  float di = rsqrtf(ri + ((ri == 0.0f) ? 1.0f : 0.0f));
  float dj = rsqrtf(rj + ((rj == 0.0f) ? 1.0f : 0.0f));
  g[base + j] = di * g[base + j] * dj;
}

// ---------------------------------------------------------------------------
// Lanczos: one workgroup (256 threads) per batch; Q-buffer + reorth in LDS.
// ---------------------------------------------------------------------------
__global__ void k_lanczos(const float* __restrict__ Lk, float* __restrict__ T,
                          float* __restrict__ Q, unsigned seed) {
  __shared__ float q[NN], qp[NN], z[NN], red[NN];
  __shared__ float Qb[NN * KK];
  __shared__ float red2[NN * KK];
  __shared__ float coef[KK], alphaA[KK], betaA[KK], validA[KK], cnorm[KK];
  __shared__ float sc0;

  int t = threadIdx.x, b = blockIdx.x;
  const float* Lb = Lk + (long long)b * NN * NN;

  for (int e = t; e < NN * KK; e += 256) Qb[e] = 0.0f;
  if (t < KK) cnorm[t] = 0.0f;

  float qi = nrand(seed + (unsigned)(b * NN + t));
  q[t] = qi; __syncthreads();
  float nrm = blockReduce256(qi * qi, red);
  q[t] = qi / sqrtf(nrm);
  qp[t] = 0.0f;
  float beta_prev = 0.0f, valid = 1.0f;
  __syncthreads();

  for (int i = 0; i < KK; ++i) {
    // z = Lk @ q
    const float* row = Lb + (long long)t * NN;
    float acc = 0.0f;
    for (int j = 0; j < NN; ++j) acc += row[j] * q[j];
    z[t] = acc; __syncthreads();

    float alpha = blockReduce256(q[t] * z[t], red);
    z[t] = z[t] - alpha * q[t] - beta_prev * qp[t];
    __syncthreads();

    // double Gram-Schmidt reorth against Qb columns
    for (int pass = 0; pass < 2; ++pass) {
      for (int k = 0; k < KK; ++k) red2[t * KK + k] = Qb[t * KK + k] * z[t];
      __syncthreads();
      for (int off = 128; off > 0; off >>= 1) {
        if (t < off)
          for (int k = 0; k < KK; ++k) red2[t * KK + k] += red2[(t + off) * KK + k];
        __syncthreads();
      }
      if (t < KK) coef[t] = red2[t] / (cnorm[t] + EPSF);
      __syncthreads();
      float sub = 0.0f;
      for (int k = 0; k < KK; ++k) sub += Qb[t * KK + k] * coef[k];
      z[t] -= sub; __syncthreads();
    }

    float beta = sqrtf(blockReduce256(z[t] * z[t], red));
    valid *= (beta >= LBF) ? 1.0f : 0.0f;
    float qn = z[t] * valid / (beta + EPSF);

    Qb[t * KK + i] = q[t];
    float cn = blockReduce256(q[t] * q[t], red);
    if (t == 0) { alphaA[i] = alpha; betaA[i] = beta; validA[i] = valid; cnorm[i] = cn; }
    qp[t] = q[t];
    __syncthreads();
    q[t] = qn;
    beta_prev = beta;
    __syncthreads();
  }

  if (t == 0) {
    float rk = 0.0f;
    for (int k = 0; k < KK; ++k) rk += validA[k];
    sc0 = rk;
  }
  __syncthreads();
  int rank = (int)sc0;

  for (int k = 0; k < KK; ++k) {
    float v = Qb[t * KK + k] * validA[k] * ((t < rank) ? 1.0f : 0.0f);
    Q[((long long)b * NN + t) * KK + k] = v;
  }
  for (int e = t; e < KK * KK; e += 256) {
    int kk = e / KK, ll = e % KK;
    float v = 0.0f;
    if (kk == ll) v = alphaA[kk] * validA[kk];
    else if (ll == kk + 1) v = betaA[kk] * validA[kk];
    else if (kk == ll + 1) v = betaA[ll] * validA[ll];
    T[(long long)b * KK * KK + e] = v;
  }
}

// ---------------------------------------------------------------------------
// T-power sequence with power-iteration spectral-radius cap. One WG / batch.
// Emits T^4, T^6, T^8 (capped) -> Tlist (B,3,K*K).
// ---------------------------------------------------------------------------
__global__ void k_tpowers(const float* __restrict__ T, float* __restrict__ Tlist,
                          unsigned seed) {
  __shared__ float Tm[KK * KK], TT[KK * KK], Tn[KK * KK], Tb[KK], Tbn[KK];
  __shared__ float sc[2];
  int t = threadIdx.x, b = blockIdx.x;

  for (int e = t; e < KK * KK; e += 256) { float v = T[b * KK * KK + e]; Tm[e] = v; TT[e] = v; }
  if (t < KK) Tb[t] = nrand(seed + (unsigned)(b * KK + t));
  __syncthreads();
  if (t == 0) {
    float s = 0.0f;
    for (int k = 0; k < KK; ++k) s += Tb[k] * Tb[k];
    sc[0] = sqrtf(s);
  }
  __syncthreads();
  if (t < KK) Tb[t] /= sc[0];
  __syncthreads();

  int sidx = 0;
  for (int ii = 0; ii < 8; ++ii) {
    if (ii == 3 || ii == 5 || ii == 7) {
      for (int e = t; e < KK * KK; e += 256) Tlist[(b * 3 + sidx) * KK * KK + e] = TT[e];
      ++sidx;
    }
    for (int e = t; e < KK * KK; e += 256) {
      int r = e / KK, c = e % KK;
      float s = 0.0f;
      for (int m = 0; m < KK; ++m) s += TT[r * KK + m] * Tm[m * KK + c];
      Tn[e] = s;
    }
    if (t < KK) {
      float s = 0.0f;
      for (int m = 0; m < KK; ++m) s += Tm[t * KK + m] * Tb[m];
      Tbn[t] = s;
    }
    __syncthreads();
    if (t == 0) {
      float num = 0.0f, den = 0.0f, nn2 = 0.0f;
      for (int k = 0; k < KK; ++k) { num += Tbn[k] * Tb[k]; den += Tb[k] * Tb[k]; nn2 += Tbn[k] * Tbn[k]; }
      float lmax = num / (den + EPSF);
      sc[0] = (fabsf(lmax) <= 1.0f) ? 1.0f : 0.0f;
      sc[1] = sqrtf(nn2) + EPSF;
    }
    __syncthreads();
    for (int e = t; e < KK * KK; e += 256) TT[e] = Tn[e] * sc[0];
    if (t < KK) Tb[t] = Tbn[t] / sc[1];
    __syncthreads();
  }
}

// triu + symmetrize DD (3,B,K,K)
__global__ void k_symmDD(const float* __restrict__ Dr, float* __restrict__ Ds) {
  int id = blockIdx.x * 256 + threadIdx.x;
  if (id >= 3 * BB * KK * KK) return;
  int s = id / (BB * KK * KK);
  int rem = id % (BB * KK * KK);
  int bb = rem / (KK * KK), e = rem % (KK * KK);
  int k = e / KK, l = e % KK;
  const float* base = Dr + (s * BB + bb) * KK * KK;
  Ds[id] = (k <= l) ? base[k * KK + l] : base[l * KK + k];
}

// final: score[b] = mean_n sigmoid(state@att)*(state@wo+bo)
__global__ void k_score(const float* __restrict__ st, const float* __restrict__ wo,
                        const float* __restrict__ bo, const float* __restrict__ aw,
                        const float* __restrict__ ab, float* __restrict__ out) {
  __shared__ float red[256];
  int t = threadIdx.x, b = blockIdx.x;
  const float* row = st + ((long long)b * NN + t) * 128;
  float y = bo[0], a = ab[0];
  for (int d = 0; d < 128; ++d) { y += row[d] * wo[d]; a += row[d] * aw[d]; }
  a = 1.0f / (1.0f + expf(-a));
  float s = blockReduce256(a * y, red);
  if (t == 0) out[b] = s / (float)NN;
}

// ---------------------------------------------------------------------------
static inline void gemm(hipStream_t st,
                        const float* A, int lda, long long sA,
                        const float* B, int ldb, long long sB,
                        float* C, int ldc, long long sC,
                        int M, int N, int K, int batch,
                        const float* bias, int relu, int tA, int tB) {
  dim3 g((N + 63) / 64, (M + 31) / 32, batch);
  k_wmma_gemm<<<g, 128, 0, st>>>(A, lda, sA, B, ldb, sB, C, ldc, sC,
                                 M, N, K, bias, relu, tA, tB);
}

extern "C" void kernel_launch(void* const* d_in, const int* in_sizes, int n_in,
                              void* d_out, int out_size, void* d_ws, size_t ws_size,
                              hipStream_t stream) {
  const int*   nf     = (const int*)d_in[0];
  const float* L      = (const float*)d_in[1];
  const float* emb_w1 = (const float*)d_in[2];
  const float* emb_b1 = (const float*)d_in[3];
  const float* emb_w2 = (const float*)d_in[4];
  const float* emb_b2 = (const float*)d_in[5];
  const float* eig_w1 = (const float*)d_in[6];
  const float* eig_b1 = (const float*)d_in[7];
  const float* eig_w2 = (const float*)d_in[8];
  const float* eig_b2 = (const float*)d_in[9];
  const float* fw0    = (const float*)d_in[10];
  const float* fb0    = (const float*)d_in[11];
  const float* fw1    = (const float*)d_in[12];
  const float* fb1    = (const float*)d_in[13];
  const float* fwo    = (const float*)d_in[14];
  const float* fbo    = (const float*)d_in[15];
  const float* aw     = (const float*)d_in[16];
  const float* ab     = (const float*)d_in[17];
  float* out = (float*)d_out;

  // workspace layout (floats); all offsets multiples of 16 floats (64B)
  float* ws = (float*)d_ws;
  float* msg    = ws;                       // 4096*1280 (h1 4096*1024 aliases here pre-msg)
  float* ledge  = msg   + 4096 * 1280;      // 4*B*N*N
  float* lk     = ledge + 4LL * BB * NN * NN;
  float* feat   = lk    + (long long)BB * NN * NN;
  float* st0    = feat  + BB * NN * DA;     // B*N*128
  float* st1    = st0   + BB * NN * 128;
  float* nrm    = st1   + BB * NN * 128;    // B*N
  float* rowpart= nrm   + BB * NN;
  float* sig    = rowpart + BB * NN;        // B (padded region follows)
  float* rowsum = sig   + 16;               // B*N
  float* T      = rowsum + BB * NN;         // B*K*K
  float* Q      = T     + BB * KK * KK;     // B*N*K
  float* Tlist  = Q     + (long long)BB * NN * KK;  // B*3*K*K
  float* eigH   = Tlist + BB * 3 * KK * KK; // 16*1024
  float* DDraw  = eigH  + 16 * 1024;        // 3*B*K*K
  float* DDsym  = DDraw + 3 * BB * KK * KK; // 3*B*K*K
  float* Pbuf   = DDsym + 3 * BB * KK * KK; // B*K*128
  float* Rbuf   = Pbuf  + BB * KK * 128;    // B*K*128
  float* h1     = msg;                      // alias (used before msg)

  // ---- embedding MLP: h1 = relu(onehot@W1+b1) == gather; feat = h1@W2+b2
  k_emb1<<<(BB * NN * 1024 + 255) / 256, 256, 0, stream>>>(nf, emb_w1, emb_b1, h1);
  gemm(stream, h1, 1024, 0, emb_w2, DA, 0, feat, DA, 0,
       BB * NN, DA, 1024, 1, emb_b2, 0, 0, 0);

  k_onehot<<<(BB * NN * DA + 255) / 256, 256, 0, stream>>>(nf, st0);
  k_slice_edge<<<(4 * BB * NN * NN + 255) / 256, 256, 0, stream>>>(L, ledge);

  // ---- graph Laplacian: Gram -> dist2 -> affinity -> sym-normalize
  gemm(stream, feat, DA, (long long)NN * DA, feat, DA, (long long)NN * DA,
       lk, NN, (long long)NN * NN, NN, NN, DA, BB, nullptr, 0, 0, 1 /*transB*/);
  k_rownorm<<<(BB * NN + 255) / 256, 256, 0, stream>>>(feat, nrm);
  k_dist<<<BB * NN, 256, 0, stream>>>(lk, nrm, rowpart);
  k_sigma<<<BB, 256, 0, stream>>>(rowpart, sig);
  k_affin<<<BB * NN, 256, 0, stream>>>(lk, L, sig, rowsum);
  k_symnorm<<<BB * NN, 256, 0, stream>>>(lk, rowsum);

  // ---- Lanczos tridiagonalization
  k_lanczos<<<BB, 256, 0, stream>>>(lk, T, Q, 0xA341316Cu);

  // ---- two message-passing layers
  for (int tt = 0; tt < 2; ++tt) {
    const int D = tt ? 128 : DA;
    const int W = 10 * D;
    const float* stateIn = tt ? st1 : st0;
    float* stateOut = tt ? st0 : st1;
    const float* fw = tt ? fw1 : fw0;
    const float* fb = tt ? fb1 : fb0;

    // spectral filters
    k_tpowers<<<BB, 256, 0, stream>>>(T, Tlist, 0xC8013EA4u + 77u * (unsigned)(tt + 1));
    for (int s = 0; s < 3; ++s) {
      gemm(stream, Tlist + s * KK * KK, 3 * KK * KK, 0,
           eig_w1 + (long long)s * KK * KK * 1024, 1024, 0,
           eigH, 1024, 0, BB, 1024, KK * KK, 1, eig_b1 + s * 1024, 1, 0, 0);
      gemm(stream, eigH, 1024, 0,
           eig_w2 + (long long)s * 1024 * KK * KK, KK * KK, 0,
           DDraw + s * BB * KK * KK, KK * KK, 0,
           BB, KK * KK, 1024, 1, eig_b2 + s * KK * KK, 0, 0, 0);
    }
    k_symmDD<<<(3 * BB * KK * KK + 255) / 256, 256, 0, stream>>>(DDraw, DDsym);

    // short diffusion: segments 0..2 (tmp = Lk @ tmp)
    gemm(stream, lk, NN, (long long)NN * NN, stateIn, D, (long long)NN * D,
         msg + 0 * D, W, (long long)NN * W, NN, D, NN, BB, nullptr, 0, 0, 0);
    gemm(stream, lk, NN, (long long)NN * NN, msg + 0 * D, W, (long long)NN * W,
         msg + 1 * D, W, (long long)NN * W, NN, D, NN, BB, nullptr, 0, 0, 0);
    gemm(stream, lk, NN, (long long)NN * NN, msg + 1 * D, W, (long long)NN * W,
         msg + 2 * D, W, (long long)NN * W, NN, D, NN, BB, nullptr, 0, 0, 0);

    // long (spectral) diffusion, factorized: Q @ (DD_s @ (Q^T @ state))
    gemm(stream, Q, KK, (long long)NN * KK, stateIn, D, (long long)NN * D,
         Pbuf, D, (long long)KK * D, KK, D, NN, BB, nullptr, 0, 1 /*transA*/, 0);
    for (int s = 0; s < 3; ++s) {
      gemm(stream, DDsym + s * BB * KK * KK, KK, (long long)KK * KK,
           Pbuf, D, (long long)KK * D, Rbuf, D, (long long)KK * D,
           KK, D, KK, BB, nullptr, 0, 0, 0);
      gemm(stream, Q, KK, (long long)NN * KK, Rbuf, D, (long long)KK * D,
           msg + (3 + s) * D, W, (long long)NN * W, NN, D, KK, BB, nullptr, 0, 0, 0);
    }

    // per-edgetype propagation: segments 6..9
    for (int e = 0; e < 4; ++e) {
      gemm(stream, ledge + (long long)e * BB * NN * NN, NN, (long long)NN * NN,
           stateIn, D, (long long)NN * D,
           msg + (6 + e) * D, W, (long long)NN * W, NN, D, NN, BB, nullptr, 0, 0, 0);
    }

    // filter MLP layer
    gemm(stream, msg, W, 0, fw, 128, 0, stateOut, 128, 0,
         BB * NN, 128, W, 1, fb, 1, 0, 0);
  }

  // ---- readout
  k_score<<<BB, 256, 0, stream>>>(st0, fwo, fbo, aw, ab, out);
}
